// ALiBiAttention_91122026152371
// MI455X (gfx1250) — matrix operand order repaired and checked by reading, not compiled, not run on hardware
//
#include <hip/hip_runtime.h>

// ---------------------------------------------------------------------------
// Types for CDNA5 WMMA (wave32): bf16 16x16x32, f32 accumulate.
// ---------------------------------------------------------------------------
typedef __attribute__((ext_vector_type(16))) __bf16 v16bf;
typedef __attribute__((ext_vector_type(8)))  __bf16 v8bf;
typedef __attribute__((ext_vector_type(8)))  float  v8f;

__device__ __forceinline__ unsigned short f2bf_bits(float f) {
    unsigned int u = __float_as_uint(f);
    u += 0x7FFFu + ((u >> 16) & 1u);           // round-to-nearest-even
    return (unsigned short)(u >> 16);
}
__device__ __forceinline__ __bf16 bits_to_bf(unsigned short b) {
    union { unsigned short u; __bf16 h; } x; x.u = b; return x.h;
}

// A-matrix (16x32, MxK) per CDNA5 ISA 7.12.2: lane m=lane%16 holds
// K runs {0..7,16..23} (lanes 0-15) / {8..15,24..31} (lanes 16-31).
__device__ __forceinline__ v16bf load_A_tile(const unsigned short* __restrict__ A,
                                             int lda, int row0, int k0, int lane) {
    const int r = lane & 15;
    const int kbase = k0 + ((lane & 16) ? 8 : 0);
    const unsigned short* p = A + (size_t)(row0 + r) * (size_t)lda + kbase;
    v8bf lo = *(const v8bf*)(p);        // K = kbase .. kbase+7
    v8bf hi = *(const v8bf*)(p + 16);   // K = kbase+16 .. kbase+23
    v16bf out;
#pragma unroll
    for (int i = 0; i < 8; ++i) { out[i] = lo[i]; out[i + 8] = hi[i]; }
    return out;
}

// B-matrix (32x16, KxN): col n=lane%16, lanes 0-15 hold K {0..7,16..23},
// lanes 16-31 hold K {8..15,24..31}. Bt is stored N-major (row n, K contiguous).
__device__ __forceinline__ v16bf load_B_tile(const unsigned short* __restrict__ Bt,
                                             int ldb, int n0, int k0, int lane) {
    const int n = lane & 15;
    const int kbase = k0 + ((lane & 16) ? 8 : 0);
    const unsigned short* p = Bt + (size_t)(n0 + n) * (size_t)ldb + kbase;
    v8bf lo = *(const v8bf*)(p);
    v8bf hi = *(const v8bf*)(p + 16);
    v16bf out;
#pragma unroll
    for (int i = 0; i < 8; ++i) { out[i] = lo[i]; out[i + 8] = hi[i]; }
    return out;
}

__device__ __forceinline__ v8f wmma_bf16(v16bf a, v16bf b, v8f c) {
    return __builtin_amdgcn_wmma_f32_16x16x32_bf16(false, a, false, b,
                                                   (short)0, c, false, false);
}

// ---------------------------------------------------------------------------
// Kernel 0a: fp32 -> bf16 elementwise
// ---------------------------------------------------------------------------
__global__ void cvt_f32_bf16_kernel(const float* __restrict__ in,
                                    unsigned short* __restrict__ out, long n) {
    long i = (long)blockIdx.x * blockDim.x + threadIdx.x;
    if (i < n) out[i] = f2bf_bits(in[i]);
}

// Kernel 0b: fp32 (R x C row-major) -> bf16 transposed (C x R)
__global__ void cvt_transpose_kernel(const float* __restrict__ in,
                                     unsigned short* __restrict__ out,
                                     int R, int C) {
    long i = (long)blockIdx.x * blockDim.x + threadIdx.x;
    long n = (long)R * C;
    if (i < n) {
        int r = (int)(i / C), c = (int)(i % C);
        out[(size_t)c * R + r] = f2bf_bits(in[i]);
    }
}

// ---------------------------------------------------------------------------
// Kernel 1: QKV GEMM.  C(M=B*T x N=3D) = Xb(MxD) @ Wqkv, weights pre-transposed.
// Wave computes a 32x64 tile (8 WMMAs / K-step; each B tile feeds 2 WMMAs).
// Block = 4 waves = 128x64. Scatters into Q (B,H,T,hd, pre-scaled 1/8),
// K (B,H,T,hd), V^T (B,H,hd,T).
// ---------------------------------------------------------------------------
__global__ void __launch_bounds__(128)
gemm_qkv_kernel(const unsigned short* __restrict__ Xb,
                const unsigned short* __restrict__ Wt,   // (3D x D), N-major
                unsigned short* __restrict__ Qb,
                unsigned short* __restrict__ Kb,
                unsigned short* __restrict__ Vt,
                int T, int D, int H, int hd) {
    const int wave = threadIdx.x >> 5;
    const int lane = threadIdx.x & 31;
    const int m0 = blockIdx.x * 128 + wave * 32;
    const int n0 = blockIdx.y * 64;

    v8f acc[2][4] = {};
    for (int k0 = 0; k0 < D; k0 += 32) {
        if (k0 + 32 < D) {   // pull next K-step A rows toward the caches
            __builtin_prefetch(Xb + (size_t)(m0 + (lane & 15)) * D + k0 + 32, 0, 1);
            __builtin_prefetch(Xb + (size_t)(m0 + 16 + (lane & 15)) * D + k0 + 32, 0, 1);
        }
        v16bf a0 = load_A_tile(Xb, D, m0,      k0, lane);
        v16bf a1 = load_A_tile(Xb, D, m0 + 16, k0, lane);
#pragma unroll
        for (int j = 0; j < 4; ++j) {
            v16bf b = load_B_tile(Wt, D, n0 + 16 * j, k0, lane);
            acc[0][j] = wmma_bf16(a0, b, acc[0][j]);
            acc[1][j] = wmma_bf16(a1, b, acc[1][j]);
        }
    }

    const float qscale = 0.125f;               // 1/sqrt(hd), hd = 64
    const int colb = lane & 15;
    const int rowadd = (lane & 16) ? 8 : 0;
#pragma unroll
    for (int g = 0; g < 2; ++g) {
#pragma unroll
        for (int j = 0; j < 4; ++j) {
            const int n = n0 + 16 * j + colb;
            const int sec = n / D;             // 0=Q, 1=K, 2=V
            const int nn = n % D;
            const int h = nn / hd, d = nn % hd;
#pragma unroll
            for (int r = 0; r < 8; ++r) {
                const int m = m0 + 16 * g + r + rowadd;
                const int bb = m / T, t = m % T;
                const float v = acc[g][j][r];
                const size_t bh = (size_t)bb * H + h;
                if (sec == 0)      Qb[(bh * T + t) * hd + d] = f2bf_bits(v * qscale);
                else if (sec == 1) Kb[(bh * T + t) * hd + d] = f2bf_bits(v);
                else               Vt[(bh * hd + d) * T + t] = f2bf_bits(v);
            }
        }
    }
}

// ---------------------------------------------------------------------------
// Kernel 2: flash attention with ALiBi. One wave = one 16-query tile of one
// (b,h). Reference (faithful to its torch quirk) masks j < i with -inf, so
// valid keys are j >= i: iterate kv chunks from the q tile to T.
// ---------------------------------------------------------------------------
__global__ void __launch_bounds__(128)
attn_kernel(const unsigned short* __restrict__ Qb,   // (B,H,T,hd) pre-scaled
            const unsigned short* __restrict__ Kb,   // (B,H,T,hd)
            const unsigned short* __restrict__ Vt,   // (B,H,hd,T)
            unsigned short* __restrict__ Ob,         // (B,T,H*hd) bf16
            int T, int H, int hd) {
    __shared__ float smem[4][16 * 32];               // per-wave P re-layout
    const int wave = threadIdx.x >> 5;
    const int lane = threadIdx.x & 31;
    const int qtiles = T / 16;
    const int task = blockIdx.x * 4 + wave;          // B*H*qtiles tasks
    const int q0 = (task % qtiles) * 16;
    const int bh = task / qtiles;
    const int h = bh % H;
    const int b = bh / H;

    const unsigned short* Q = Qb + (size_t)bh * T * hd;
    const unsigned short* K = Kb + (size_t)bh * T * hd;
    const unsigned short* V = Vt + (size_t)bh * hd * T;
    const float slope = exp2f(-0.5f * (float)(h + 1));   // ALiBi, H=16
    const float NEG = -3.0e38f;

    v16bf aq0 = load_A_tile(Q, hd, q0, 0, lane);
    v16bf aq1 = load_A_tile(Q, hd, q0, 32, lane);

    v8f o[4] = {};                                   // 16 x 64 f32 accum
    float mrow[8], lrow[8];
#pragma unroll
    for (int r = 0; r < 8; ++r) { mrow[r] = NEG; lrow[r] = 0.0f; }

    const int rowadd = (lane & 16) ? 8 : 0;
    const int col = lane & 15;
    float* S = smem[wave];

    for (int kv0 = (q0 & ~31); kv0 < T; kv0 += 32) {
        // ---- S = Q @ K^T for a 16x32 chunk (two 16x16 tiles, K-dim = hd) --
        v8f s[2] = {};
#pragma unroll
        for (int half = 0; half < 2; ++half) {
            v16bf bk0 = load_B_tile(K, hd, kv0 + 16 * half, 0, lane);
            s[half] = wmma_bf16(aq0, bk0, s[half]);
            v16bf bk1 = load_B_tile(K, hd, kv0 + 16 * half, 32, lane);
            s[half] = wmma_bf16(aq1, bk1, s[half]);
        }
        // ---- ALiBi bias + (inverted) causal mask, chunk row max -----------
        float smax[8];
#pragma unroll
        for (int r = 0; r < 8; ++r) smax[r] = NEG;
#pragma unroll
        for (int half = 0; half < 2; ++half) {
            const int j = kv0 + 16 * half + col;
#pragma unroll
            for (int r = 0; r < 8; ++r) {
                const int i = q0 + r + rowadd;
                const int dd = j - i;
                const float val = (dd < 0) ? NEG : fmaf(-slope, (float)dd, s[half][r]);
                s[half][r] = val;
                smax[r] = fmaxf(smax[r], val);
            }
        }
#pragma unroll
        for (int r = 0; r < 8; ++r) {                // row max across 16 lanes
            float v = smax[r];
            v = fmaxf(v, __shfl_xor(v, 1, 16));
            v = fmaxf(v, __shfl_xor(v, 2, 16));
            v = fmaxf(v, __shfl_xor(v, 4, 16));
            v = fmaxf(v, __shfl_xor(v, 8, 16));
            smax[r] = v;
        }
        // ---- online softmax update; P -> LDS (row-major f32) --------------
        float psum[8];
#pragma unroll
        for (int r = 0; r < 8; ++r) {
            const float mnew = fmaxf(mrow[r], smax[r]);
            const float corr = __expf(mrow[r] - mnew);
            mrow[r] = mnew;
            lrow[r] = lrow[r] * corr;
#pragma unroll
            for (int t = 0; t < 4; ++t) o[t][r] = o[t][r] * corr;
            float ps = 0.0f;
#pragma unroll
            for (int half = 0; half < 2; ++half) {
                const float p = __expf(s[half][r] - mnew);
                ps += p;
                S[(r + rowadd) * 32 + 16 * half + col] = p;
            }
            psum[r] = ps;
        }
#pragma unroll
        for (int r = 0; r < 8; ++r) {                // row sum across 16 lanes
            float v = psum[r];
            v += __shfl_xor(v, 1, 16);
            v += __shfl_xor(v, 2, 16);
            v += __shfl_xor(v, 4, 16);
            v += __shfl_xor(v, 8, 16);
            lrow[r] += v;
        }
        // ---- re-layout P (C-format -> A-format bf16) via LDS --------------
        const int pr = lane & 15;
        const int kb = (lane & 16) ? 8 : 0;
        v16bf pa;
#pragma unroll
        for (int i = 0; i < 8; ++i) {
            pa[i]     = bits_to_bf(f2bf_bits(S[pr * 32 + kb + i]));
            pa[i + 8] = bits_to_bf(f2bf_bits(S[pr * 32 + kb + 16 + i]));
        }
        // ---- O += P @ V (K-dim = 32 kv positions, 4 d-tiles) --------------
#pragma unroll
        for (int t = 0; t < 4; ++t) {
            v16bf bv = load_B_tile(V, T, 16 * t, kv0, lane);
            o[t] = wmma_bf16(pa, bv, o[t]);
        }
    }

    // ---- normalize and store bf16 attention output (B,T,H*hd) -------------
    const int Dm = H * hd;
#pragma unroll
    for (int r = 0; r < 8; ++r) {
        const float inv = 1.0f / lrow[r];
        const int t = q0 + r + rowadd;
#pragma unroll
        for (int j = 0; j < 4; ++j) {
            const float v = o[j][r] * inv;
            Ob[((size_t)b * T + t) * Dm + h * hd + 16 * j + col] = f2bf_bits(v);
        }
    }
}

// ---------------------------------------------------------------------------
// Kernel 3: output GEMM.  out(M x N f32) = Ob(M x K bf16) @ Wout (N-major bf16)
// Wave computes 32x64 (8 WMMAs / K-step), block = 4 waves = 128x64.
// ---------------------------------------------------------------------------
__global__ void __launch_bounds__(128)
gemm_out_kernel(const unsigned short* __restrict__ Ab,
                const unsigned short* __restrict__ Wt,   // (N x K)
                float* __restrict__ Cout, int N, int Kd) {
    const int wave = threadIdx.x >> 5;
    const int lane = threadIdx.x & 31;
    const int m0 = blockIdx.x * 128 + wave * 32;
    const int n0 = blockIdx.y * 64;

    v8f acc[2][4] = {};
    for (int k0 = 0; k0 < Kd; k0 += 32) {
        if (k0 + 32 < Kd) {
            __builtin_prefetch(Ab + (size_t)(m0 + (lane & 15)) * Kd + k0 + 32, 0, 1);
            __builtin_prefetch(Ab + (size_t)(m0 + 16 + (lane & 15)) * Kd + k0 + 32, 0, 1);
        }
        v16bf a0 = load_A_tile(Ab, Kd, m0,      k0, lane);
        v16bf a1 = load_A_tile(Ab, Kd, m0 + 16, k0, lane);
#pragma unroll
        for (int j = 0; j < 4; ++j) {
            v16bf bm = load_B_tile(Wt, Kd, n0 + 16 * j, k0, lane);
            acc[0][j] = wmma_bf16(a0, bm, acc[0][j]);
            acc[1][j] = wmma_bf16(a1, bm, acc[1][j]);
        }
    }
    const int colb = lane & 15;
    const int rowadd = (lane & 16) ? 8 : 0;
#pragma unroll
    for (int g = 0; g < 2; ++g) {
#pragma unroll
        for (int j = 0; j < 4; ++j) {
            const int n = n0 + 16 * j + colb;
#pragma unroll
            for (int r = 0; r < 8; ++r)
                Cout[(size_t)(m0 + 16 * g + r + rowadd) * N + n] = acc[g][j][r];
        }
    }
}

// ---------------------------------------------------------------------------
extern "C" void kernel_launch(void* const* d_in, const int* in_sizes, int n_in,
                              void* d_out, int out_size, void* d_ws, size_t ws_size,
                              hipStream_t stream) {
    (void)in_sizes; (void)n_in; (void)out_size; (void)ws_size;
    const float* x    = (const float*)d_in[0];   // (B,T,D) f32
    const float* Wqkv = (const float*)d_in[1];   // (D,3D)  f32
    const float* Wout = (const float*)d_in[2];   // (D,D)   f32
    float* out = (float*)d_out;                  // (B,T,D) f32

    const int B = 2, T = 2048, D = 1024, H = 16, hd = 64;
    const int M = B * T;                         // 4096
    const int N1 = 3 * D;                        // 3072

    // Workspace carve-up (all bf16 stored as u16 bit patterns): ~48 MiB total
    unsigned short* p = (unsigned short*)d_ws;
    unsigned short* Xb    = p; p += (size_t)M * D;     // x in bf16
    unsigned short* Wqkvt = p; p += (size_t)N1 * D;    // W_qkv^T (N-major)
    unsigned short* Woutt = p; p += (size_t)D * D;     // W_out^T (N-major)
    unsigned short* Qb    = p; p += (size_t)M * D;     // (B,H,T,hd) *1/8
    unsigned short* Kb    = p; p += (size_t)M * D;     // (B,H,T,hd)
    unsigned short* Vt    = p; p += (size_t)M * D;     // (B,H,hd,T)
    unsigned short* Ob    = p; p += (size_t)M * D;     // (B,T,D) bf16

    {   // conversions / transposes
        long n = (long)M * D;
        cvt_f32_bf16_kernel<<<(int)((n + 255) / 256), 256, 0, stream>>>(x, Xb, n);
        long nw = (long)D * N1;
        cvt_transpose_kernel<<<(int)((nw + 255) / 256), 256, 0, stream>>>(Wqkv, Wqkvt, D, N1);
        long no = (long)D * D;
        cvt_transpose_kernel<<<(int)((no + 255) / 256), 256, 0, stream>>>(Wout, Woutt, D, D);
    }

    dim3 g1(M / 128, N1 / 64);                   // 32 x 48 blocks
    gemm_qkv_kernel<<<g1, 128, 0, stream>>>(Xb, Wqkvt, Qb, Kb, Vt, T, D, H, hd);

    const int tasks = B * H * (T / 16);          // 4096 wave tasks
    attn_kernel<<<tasks / 4, 128, 0, stream>>>(Qb, Kb, Vt, Ob, T, H, hd);

    dim3 g2(M / 128, D / 64);                    // 32 x 16 blocks
    gemm_out_kernel<<<g2, 128, 0, stream>>>(Ob, Woutt, out, D, D);
}
